// PILayer_12532714569873
// MI455X (gfx1250) — compile-verified, matrix operand order checked by reading, not run on hardware
//
#include <hip/hip_runtime.h>
#include <hip/hip_bf16.h>

typedef __attribute__((ext_vector_type(16))) _Float16 v16h;
typedef __attribute__((ext_vector_type(8)))  float    v8f;
typedef __attribute__((ext_vector_type(4)))  float    v4f;

#define D_IN  128
#define D_HID 256
#define D_OUT 128

__device__ __forceinline__ float fast_tanh(float x) {
#if __has_builtin(__builtin_amdgcn_tanhf)
    return __builtin_amdgcn_tanhf(x);
#else
    float xc = fminf(fmaxf(x, -10.0f), 10.0f);
    float t  = __builtin_amdgcn_exp2f(xc * 2.8853900817779268f);
    return (t - 1.0f) * __builtin_amdgcn_rcpf(t + 1.0f);
#endif
}

// ---------------------------------------------------------------------------
// Prep: permute W1 [128][256] f32 -> f16 fragments (used as GEMM1's A = W1^T).
// Fragment frag = nt*4 + kt (nt: hidden tile 0..15, kt: K tile over D_IN 0..3)
// element e: run=e>>3, pos=e&7, half=lane>>4
//   k_in  = kt*32 + run*16 + half*8 + pos
//   n_hid = nt*16 + (lane&15)
//   value = W1[k_in][n_hid]
// ---------------------------------------------------------------------------
__global__ void prep_w1(const float* __restrict__ W1, _Float16* __restrict__ W1p) {
    int t = blockIdx.x * blockDim.x + threadIdx.x;      // 0..32767
    if (t >= 64 * 32 * 16) return;
    int e    = t & 15;
    int lane = (t >> 4) & 31;
    int frag = t >> 9;
    int kt = frag & 3, nt = frag >> 2;
    int half = lane >> 4, run = e >> 3, pos = e & 7;
    int k = kt * 32 + run * 16 + half * 8 + pos;
    int n = nt * 16 + (lane & 15);
    W1p[t] = (_Float16)W1[k * D_HID + n];
}

// W2 [256][128] -> f16 fragments (used as GEMM2's B). frag = nt2*8 + kt2.
__global__ void prep_w2(const float* __restrict__ W2, _Float16* __restrict__ W2p) {
    int t = blockIdx.x * blockDim.x + threadIdx.x;      // 0..32767
    if (t >= 64 * 32 * 16) return;
    int e    = t & 15;
    int lane = (t >> 4) & 31;
    int frag = t >> 9;
    int kt = frag & 7, nt = frag >> 3;
    int half = lane >> 4, run = e >> 3, pos = e & 7;
    int k = kt * 32 + run * 16 + half * 8 + pos;
    int n = nt * 16 + (lane & 15);
    W2p[t] = (_Float16)W2[k * D_OUT + n];
}

// ---------------------------------------------------------------------------
// Main: 8 waves/block, each wave owns a 16-edge tile.
// GEMM1 (transposed): h^T = W1^T(A, LDS) x inter^T(B, regs) -> D in regs,
// lane holds edge row m, VGPRs hold hidden indices -> feeds GEMM2 A directly.
// GEMM2: out = h(A, regs) x W2(B, LDS).
// LDS: sW1 (64KB) | sW2 (64KB) = 128KB -> 2 blocks / WGP.
// ---------------------------------------------------------------------------
__global__ void __launch_bounds__(256)
pilayer_main(const int* __restrict__ idx_i, const int* __restrict__ idx_j,
             const float* __restrict__ prop, const float* __restrict__ basis,
             const _Float16* __restrict__ W1p, const float* __restrict__ b1,
             const _Float16* __restrict__ W2p, const float* __restrict__ b2,
             float* __restrict__ out, int E) {
    extern __shared__ uint4 smem_u4[];
    _Float16* sW1 = (_Float16*)smem_u4;           // 32768 halves
    _Float16* sW2 = sW1 + 32768;                  // 32768 halves

    // ---- stage weights into LDS (128KB) ----
    {
        const uint4* s1 = (const uint4*)W1p;
        const uint4* s2 = (const uint4*)W2p;
        uint4* d1 = (uint4*)sW1;
        uint4* d2 = (uint4*)sW2;
        #pragma unroll 4
        for (int i = threadIdx.x; i < 4096; i += 256) {
            d1[i] = s1[i];
            d2[i] = s2[i];
        }
    }
    __syncthreads();

    const int wave    = threadIdx.x >> 5;
    const int lane    = threadIdx.x & 31;
    const int tile    = blockIdx.x * 8 + wave;
    const int ntiles  = (E + 15) >> 4;
    if (tile >= ntiles) return;                   // uniform per wave: EXEC stays all-1

    const int m       = lane & 15;                // edge row handled by this lane
    const int halfsel = lane >> 4;                // K-half selector (ISA A/B layout)
    const int base    = tile * 16;
    int e  = base + m;
    int eg = e < E ? e : E - 1;

    const int i = idx_i[eg];
    const int j = idx_j[eg];
    const float* pi = prop  + (long)i  * D_IN;
    const float* pj = prop  + (long)j  * D_IN;
    const float* bs = basis + (long)eg * D_IN;

    // ---- build inter^T B-fragments: inter = prop[i] + prop[j] + basis ----
    v16h binter[4];
    #pragma unroll
    for (int kt = 0; kt < 4; ++kt) {
        #pragma unroll
        for (int run = 0; run < 2; ++run) {
            int koff = kt * 32 + run * 16 + halfsel * 8;
            v4f s0 = *(const v4f*)(pi + koff)     + *(const v4f*)(pj + koff)
                   + *(const v4f*)(bs + koff);
            v4f s1 = *(const v4f*)(pi + koff + 4) + *(const v4f*)(pj + koff + 4)
                   + *(const v4f*)(bs + koff + 4);
            #pragma unroll
            for (int c = 0; c < 4; ++c) {
                binter[kt][run * 8 + c]     = (_Float16)s0[c];
                binter[kt][run * 8 + 4 + c] = (_Float16)s1[c];
            }
        }
    }

    // ---- GEMM1: h^T tiles; nt-pair (2t, 2t+1) forms GEMM2 A-fragment t ----
    v16h afrag[8];
    #pragma unroll
    for (int t = 0; t < 8; ++t) {
        v8f accL = {0.f, 0.f, 0.f, 0.f, 0.f, 0.f, 0.f, 0.f};
        v8f accH = {0.f, 0.f, 0.f, 0.f, 0.f, 0.f, 0.f, 0.f};
        #pragma unroll
        for (int kt = 0; kt < 4; ++kt) {
            v16h wa = *(const v16h*)(sW1 + (((2 * t) * 4 + kt) * 32 + lane) * 16);
            accL = __builtin_amdgcn_wmma_f32_16x16x32_f16(
                       false, wa, false, binter[kt], (short)0, accL, false, false);
        }
        #pragma unroll
        for (int kt = 0; kt < 4; ++kt) {
            v16h wa = *(const v16h*)(sW1 + (((2 * t + 1) * 4 + kt) * 32 + lane) * 16);
            accH = __builtin_amdgcn_wmma_f32_16x16x32_f16(
                       false, wa, false, binter[kt], (short)0, accH, false, false);
        }
        // bias over hidden index = 32t + run*16 + 8*halfsel + r  (aligned v4f loads)
        v4f bl0 = *(const v4f*)(b1 + 32 * t + 8 * halfsel);
        v4f bl1 = *(const v4f*)(b1 + 32 * t + 8 * halfsel + 4);
        v4f bh0 = *(const v4f*)(b1 + 32 * t + 16 + 8 * halfsel);
        v4f bh1 = *(const v4f*)(b1 + 32 * t + 16 + 8 * halfsel + 4);
        #pragma unroll
        for (int r = 0; r < 4; ++r) {
            afrag[t][r]      = (_Float16)fast_tanh(accL[r]     + bl0[r]);
            afrag[t][4 + r]  = (_Float16)fast_tanh(accL[4 + r] + bl1[r]);
            afrag[t][8 + r]  = (_Float16)fast_tanh(accH[r]     + bh0[r]);
            afrag[t][12 + r] = (_Float16)fast_tanh(accH[4 + r] + bh1[r]);
        }
    }

    // ---- GEMM2: out = h x W2 ----
    #pragma unroll 1
    for (int nt2 = 0; nt2 < 8; ++nt2) {
        v8f acc = {0.f, 0.f, 0.f, 0.f, 0.f, 0.f, 0.f, 0.f};
        #pragma unroll
        for (int kt2 = 0; kt2 < 8; ++kt2) {
            v16h wb = *(const v16h*)(sW2 + ((nt2 * 8 + kt2) * 32 + lane) * 16);
            acc = __builtin_amdgcn_wmma_f32_16x16x32_f16(
                      false, afrag[kt2], false, wb, (short)0, acc, false, false);
        }
        float bias = b2[nt2 * 16 + (lane & 15)];
        #pragma unroll
        for (int r = 0; r < 8; ++r) {
            int er = base + r + 8 * halfsel;
            if (er < E)
                out[(long)er * D_OUT + nt2 * 16 + (lane & 15)] = acc[r] + bias;
        }
    }
}

// ---------------------------------------------------------------------------
extern "C" void kernel_launch(void* const* d_in, const int* in_sizes, int n_in,
                              void* d_out, int out_size, void* d_ws, size_t ws_size,
                              hipStream_t stream) {
    const int*   idx_i = (const int*)d_in[0];
    const int*   idx_j = (const int*)d_in[1];
    const float* prop  = (const float*)d_in[2];
    const float* basis = (const float*)d_in[3];
    const float* W1    = (const float*)d_in[4];
    const float* b1    = (const float*)d_in[5];
    const float* W2    = (const float*)d_in[6];
    const float* b2    = (const float*)d_in[7];
    float*       out   = (float*)d_out;
    const int E = in_sizes[0];

    _Float16* W1p = (_Float16*)d_ws;
    _Float16* W2p = (_Float16*)((char*)d_ws + 32768 * sizeof(_Float16));

    prep_w1<<<128, 256, 0, stream>>>(W1, W1p);
    prep_w2<<<128, 256, 0, stream>>>(W2, W2p);

    int ntiles = (E + 15) / 16;
    int blocks = (ntiles + 7) / 8;
    size_t shmem = (size_t)(32768 + 32768) * sizeof(_Float16); // 128KB
    pilayer_main<<<blocks, 256, shmem, stream>>>(idx_i, idx_j, prop, basis,
                                                 W1p, b1, W2p, b2, out, E);
}
